// EncoderSimilarity_77171972374988
// MI455X (gfx1250) — matrix operand order repaired and checked by reading, not compiled, not run on hardware
//
#include <hip/hip_runtime.h>
#include <hip/hip_bf16.h>

// ---------------------------------------------------------------------------
// EncoderSimilarity for MI455X (gfx1250): bf16 WMMA GEMMs with LDS-shared
// B-operands staged via GLOBAL_LOAD_ASYNC_TO_LDS (double buffered, ASYNCcnt),
// f32 VALU epilogues.
// Shapes: B=64, V=36, K=1024, Nc=64, L=48, H=9
// ---------------------------------------------------------------------------

typedef __attribute__((ext_vector_type(16))) __bf16 v16bf;
typedef __attribute__((ext_vector_type(8)))  float  v8f;

#define KDIM 1024
#define BDIM 64
#define VDIM 36
#define NCAP 64
#define LDIM 48
#define HOUT 9
#define N1   (HOUT * LDIM)   // 432
#define M1   (BDIM * VDIM)   // 2304
#define ROWS (NCAP * BDIM)   // 4096

__device__ __forceinline__ v8f wmma_bf16(v16bf a, v16bf b, v8f c) {
  return __builtin_amdgcn_wmma_f32_16x16x32_bf16(false, a, false, b, (short)0, c,
                                                 false, false);
}

// A-fragment (16x32 MxK, row-major bf16, ld): lanes 0-15 row=lane, K {0..15};
// lanes 16-31 row=lane-16, K {16..31}.
__device__ __forceinline__ v16bf load_a_frag(const __bf16* base, int ld) {
  int lane = threadIdx.x & 31;
  int row  = lane & 15;
  int kb   = (lane >> 4) << 4;
  const __bf16* p = base + (size_t)row * ld + kb;
  v16bf a;
#pragma unroll
  for (int i = 0; i < 8; ++i) { a[i] = p[i]; a[8 + i] = p[8 + i]; }
  return a;
}

// B-fragment (32x16 KxN) from an n-major [n][32k] store (contiguous 16 per lane).
__device__ __forceinline__ v16bf load_b_lane(const __bf16* lanep) {
  v16bf b;
#pragma unroll
  for (int i = 0; i < 16; ++i) b[i] = lanep[i];
  return b;
}

// 16-byte async copy: global -> LDS, tracked by ASYNCcnt. IOFFSET applies to
// both the LDS destination and the global source address (ISA 08_async_tensor §4.4).
__device__ __forceinline__ void async_copy16(unsigned lds_addr, const void* gptr) {
  asm volatile("global_load_async_to_lds_b128 %0, %1, off"
               :: "v"(lds_addr), "v"(gptr) : "memory");
}
__device__ __forceinline__ void async_copy16_off16(unsigned lds_addr, const void* gptr) {
  asm volatile("global_load_async_to_lds_b128 %0, %1, off offset:16"
               :: "v"(lds_addr), "v"(gptr) : "memory");
}
__device__ __forceinline__ void wait_async0() {
  asm volatile("s_wait_asynccnt 0x0" ::: "memory");
}
__device__ __forceinline__ unsigned lds_addr_of(const void* p) {
  return (unsigned)(uintptr_t)p;   // flat LDS aperture: addr[31:0] = LDS offset
}

// ------------------------------- converters --------------------------------

__global__ void k_cvt_bf16(const float* __restrict__ src, __bf16* __restrict__ dst, int n) {
  int i = blockIdx.x * blockDim.x + threadIdx.x;
  if (i < n) dst[i] = (__bf16)src[i];
}

// capT[j][k][q] = bf16(cap[j][q][k])  -- q-contiguous for k_lall B-fragments
__global__ void k_capT(const float* __restrict__ cap, __bf16* __restrict__ capT) {
  int idx = blockIdx.x * blockDim.x + threadIdx.x;
  if (idx >= NCAP * KDIM * LDIM) return;
  int j = idx / (KDIM * LDIM);
  int r = idx % (KDIM * LDIM);
  int k = r / LDIM, q = r % LDIM;
  capT[idx] = (__bf16)cap[((size_t)j * LDIM + q) * KDIM + k];
}

// hcap[n][k] = bf16(h_mat[h][k] * cap[j][q][k]),  n = h*48+q  (k-contiguous)
__global__ void k_hcap(const float* __restrict__ cap, const float* __restrict__ hmat,
                       __bf16* __restrict__ hcap, int j) {
  int idx = blockIdx.x * blockDim.x + threadIdx.x;   // < 432*1024
  int n = idx >> 10, k = idx & (KDIM - 1);
  int h = n / LDIM, q = n % LDIM;
  hcap[idx] = (__bf16)(hmat[(size_t)h * KDIM + k] * cap[((size_t)j * LDIM + q) * KDIM + k]);
}

// ------------------------- stage 1: bilinear logits ------------------------
// logits[(b*36+v)*432 + n] = sum_k imgb[bv,k] * hcap[n,k] + h_bias[n/48]
// grid(9 ngrp, 18 mgrp) block(256 = 8 waves). Workgroup tile: 128 rows x 48 cols.
// B strip (48x32/K-step) staged async into double-buffered LDS, shared by 8 waves.
__global__ void k_logits(const __bf16* __restrict__ imgb,
                         const __bf16* __restrict__ hcap,
                         const float*  __restrict__ hbias,
                         float* __restrict__ logits) {
  __shared__ __bf16 Bs[2][48][32];                 // 6 KB
  int ng   = blockIdx.x;                           // 48-col strip
  int mgrp = blockIdx.y;
  int w    = threadIdx.x >> 5;                     // wave id -> mtile
  int lane = threadIdx.x & 31;
  int colL = lane & 15;
  int kbB  = (lane >> 4) << 4;
  int mb   = (lane >> 4) << 3;
  int mtile = mgrp * 8 + w;
  int n0    = ng * 48;

  // staging role: 192 threads async-copy 16B each (48 cols x 32 halfs = 3 KB)
  int st   = threadIdx.x;
  int scol = st >> 2, sseg = (st & 3) * 8;

  auto stage = [&](int buf, int kk) {
    if (st < 192) {
      async_copy16(lds_addr_of(&Bs[buf][scol][sseg]),
                   hcap + (size_t)(n0 + scol) * KDIM + kk + sseg);
    }
  };

  v8f acc[3] = {};
  const __bf16* Ab = imgb + (size_t)mtile * 16 * KDIM;

  stage(0, 0);
  wait_async0();
  __syncthreads();
  for (int ks = 0; ks < KDIM / 32; ++ks) {
    int kk  = ks * 32;
    int cur = ks & 1;
    if (ks + 1 < KDIM / 32) stage(cur ^ 1, kk + 32);
    v16bf a = load_a_frag(Ab + kk, KDIM);
#pragma unroll
    for (int t = 0; t < 3; ++t) {
      v16bf b = load_b_lane(&Bs[cur][t * 16 + colL][kbB]);
      acc[t] = wmma_bf16(a, b, acc[t]);
    }
    wait_async0();
    __syncthreads();
  }

#pragma unroll
  for (int t = 0; t < 3; ++t) {
    int n = n0 + t * 16 + colL;
    float bias = hbias[n / LDIM];
#pragma unroll
    for (int r = 0; r < 8; ++r) {
      int row = mtile * 16 + mb + r;
      logits[(size_t)row * N1 + n] = acc[t][r] + bias;
    }
  }
}

// ------------------------- stage 2: masked softmax -------------------------
__global__ void k_softmax(float* __restrict__ logits, const int* __restrict__ lens, int j) {
  __shared__ float red[256];
  int bh = blockIdx.x, b = bh / HOUT, h = bh % HOUT;
  int len = lens[j];
  int tid = threadIdx.x;

  float mx = -3.4e38f;
  for (int i = tid; i < VDIM * LDIM; i += 256) {
    int v = i / LDIM, q = i % LDIM;
    if (q < len) mx = fmaxf(mx, logits[(size_t)(b * VDIM + v) * N1 + h * LDIM + q]);
  }
  red[tid] = mx; __syncthreads();
  for (int s = 128; s > 0; s >>= 1) { if (tid < s) red[tid] = fmaxf(red[tid], red[tid + s]); __syncthreads(); }
  mx = red[0]; __syncthreads();

  float sum = 0.f;
  for (int i = tid; i < VDIM * LDIM; i += 256) {
    int v = i / LDIM, q = i % LDIM;
    size_t idx = (size_t)(b * VDIM + v) * N1 + h * LDIM + q;
    float e = (q < len) ? __expf(logits[idx] - mx) : 0.f;
    logits[idx] = e;
    sum += e;
  }
  red[tid] = sum; __syncthreads();
  for (int s = 128; s > 0; s >>= 1) { if (tid < s) red[tid] += red[tid + s]; __syncthreads(); }
  float inv = 1.f / red[0]; __syncthreads();

  for (int i = tid; i < VDIM * LDIM; i += 256) {
    int v = i / LDIM, q = i % LDIM;
    size_t idx = (size_t)(b * VDIM + v) * N1 + h * LDIM + q;
    logits[idx] *= inv;
  }
}

// -------------------- stage 3: attended features l_all ---------------------
// One wave per (b,h): keeps all 6 A-fragments (p rows, 48x64 padded) in regs,
// loops over all 64 feature tiles. Block = 9 waves (h=0..8) at fixed b so
// img/capT re-reads hit the WGP$.  t = p(48x64) x capT(64x16); then fused
// v-reduction against f32 img; cross-half fold via shfl_xor(16).
__global__ void k_lall(const float* __restrict__ p,
                       const __bf16* __restrict__ capT,
                       const float*  __restrict__ imgf,
                       __bf16* __restrict__ lall, int j) {
  int b = blockIdx.x;
  int h = threadIdx.x >> 5;            // wave id == head
  int lane = threadIdx.x & 31;
  int colL = lane & 15;
  int kbA  = (lane >> 4) << 4;
  int kbB  = (lane >> 4) << 4;
  int mb   = (lane >> 4) << 3;

  // ---- load A fragments once: m in 0..2 (v rows), kk in {0,32} (q) ----
  v16bf afr[3][2];
#pragma unroll
  for (int m = 0; m < 3; ++m) {
    int vrow = m * 16 + colL;
    const float* prow = p + (size_t)(b * VDIM + (vrow < VDIM ? vrow : 0)) * N1 + h * LDIM;
#pragma unroll
    for (int kx = 0; kx < 2; ++kx) {
      int kk = kx * 32;
      bool va0 = (vrow < VDIM) && (kk + kbA)      < LDIM;  // starts {0,16,32,48}
      bool va1 = (vrow < VDIM) && (kk + kbA + 8)  < LDIM;  // starts {8,24,40,56}
      v16bf a;
#pragma unroll
      for (int i = 0; i < 8; ++i) {
        a[i]     = va0 ? (__bf16)prow[kk + kbA + i]      : (__bf16)0.f;
        a[8 + i] = va1 ? (__bf16)prow[kk + kbA + 8 + i] : (__bf16)0.f;
      }
      afr[m][kx] = a;
    }
  }

  const __bf16* capTj = capT + (size_t)j * KDIM * LDIM;
  for (int nt = 0; nt < KDIM / 16; ++nt) {
    int kf = nt * 16 + colL;
    // B fragments for this feature tile (shared across the 3 M-tiles)
    v16bf bfr[2];
#pragma unroll
    for (int kx = 0; kx < 2; ++kx) {
      int kk = kx * 32;
      bool vb = (kk + kbB) < LDIM;                 // starts {0,16,32,48}
      const __bf16* bp = capTj + (size_t)kf * LDIM + kk + kbB;
      v16bf bb;
#pragma unroll
      for (int i = 0; i < 16; ++i) bb[i] = vb ? bp[i] : (__bf16)0.f;
      bfr[kx] = bb;
    }
    float acc = 0.f;
#pragma unroll
    for (int m = 0; m < 3; ++m) {
      v8f c = {};
      c = wmma_bf16(afr[m][0], bfr[0], c);
      c = wmma_bf16(afr[m][1], bfr[1], c);
#pragma unroll
      for (int r = 0; r < 8; ++r) {
        int v = m * 16 + mb + r;
        if (v < VDIM) acc += imgf[((size_t)b * VDIM + v) * KDIM + kf] * c[r];
      }
    }
    acc += __shfl_xor(acc, 16);
    if (lane < 16)
      lall[((size_t)h * ROWS + (size_t)j * BDIM + b) * KDIM + kf] = (__bf16)acc;
  }
}

// --------------------- stage 4: batched gated-step GEMM --------------------
// out[r,o] = sum_k A[r,k]*W[o,k] + bias[o].  grid(8 ngrp, 32 mgrp) block(256).
// Workgroup tile 128x128; B tile (128x32/K-step, 8KB) staged async into
// double-buffered LDS shared by 8 M-waves; A fragments from global (b128 pairs).
__global__ void k_gemm_bias(const __bf16* __restrict__ A,
                            const __bf16* __restrict__ Wb,
                            const float*  __restrict__ bias,
                            float* __restrict__ out) {
  __shared__ __bf16 Bs[2][128][32];                // 16 KB
  int ngrp = blockIdx.x;
  int mgrp = blockIdx.y;
  int w    = threadIdx.x >> 5;
  int lane = threadIdx.x & 31;
  int colL = lane & 15;
  int mb   = (lane >> 4) << 3;
  int kbB  = (lane >> 4) << 4;
  int mtile = mgrp * 8 + w;
  int ncol0 = ngrp * 128;

  // staging role: 256 threads async-copy 32B each (128 cols x 32 halfs = 8 KB)
  int st = threadIdx.x;
  int scol = st >> 1, sseg = (st & 1) * 16;

  auto stage = [&](int buf, int kk) {
    unsigned laddr = lds_addr_of(&Bs[buf][scol][sseg]);
    const __bf16* g = Wb + (size_t)(ncol0 + scol) * KDIM + kk + sseg;
    async_copy16(laddr, g);
    async_copy16_off16(laddr, g);    // IOFFSET advances both LDS and global
  };

  v8f acc[8] = {};
  const __bf16* Ab = A + (size_t)mtile * 16 * KDIM;

  stage(0, 0);
  wait_async0();
  __syncthreads();
  for (int ks = 0; ks < KDIM / 32; ++ks) {
    int kk  = ks * 32;
    int cur = ks & 1;
    if (ks + 1 < KDIM / 32) stage(cur ^ 1, kk + 32);
    v16bf a = load_a_frag(Ab + kk, KDIM);
    if (kk + 32 < KDIM)
      __builtin_prefetch(Ab + (size_t)(lane & 15) * KDIM + kk + 32, 0, 0);
#pragma unroll
    for (int t = 0; t < 8; ++t) {
      v16bf b = load_b_lane(&Bs[cur][t * 16 + colL][kbB]);
      acc[t] = wmma_bf16(a, b, acc[t]);
    }
    wait_async0();
    __syncthreads();
  }

#pragma unroll
  for (int t = 0; t < 8; ++t) {
    int col = ncol0 + t * 16 + colL;
    float bi = bias[col];
#pragma unroll
    for (int r = 0; r < 8; ++r) {
      int row = mtile * 16 + mb + r;
      out[(size_t)row * KDIM + col] = acc[t][r] + bi;
    }
  }
}

// elementwise pre: G = l1 + tanh(l2) -> bf16; also stash l2 as bf16 (W3 operand)
__global__ void k_gated_pre(const __bf16* __restrict__ l1b,
                            const float*  __restrict__ l2f,
                            const __bf16* __restrict__ l2bsrc, int l2_is_b,
                            __bf16* __restrict__ Gb, __bf16* __restrict__ l2b) {
  size_t i = (size_t)blockIdx.x * blockDim.x + threadIdx.x;
  if (i >= (size_t)ROWS * KDIM) return;
  float l1 = (float)l1b[i];
  float l2 = l2_is_b ? (float)l2bsrc[i] : l2f[i];
  Gb[i]  = (__bf16)(l1 + tanhf(l2));
  l2b[i] = (__bf16)l2;
}

// elementwise post + row l2norm:
// a = tanh(l1)*sigmoid(o1)+l1 ; b = o3*sigmoid(o2)+l2 ; lts = l2norm(a+b)
__global__ void k_gated_post(const __bf16* __restrict__ l1b,
                             const float*  __restrict__ l2f,
                             const __bf16* __restrict__ l2bsrc, int l2_is_b,
                             const float* __restrict__ o1,
                             const float* __restrict__ o2,
                             const float* __restrict__ o3,
                             float* __restrict__ lts) {
  __shared__ float red[256];
  int row = blockIdx.x, tid = threadIdx.x;
  float sv[4], ss = 0.f;
#pragma unroll
  for (int u = 0; u < 4; ++u) {
    size_t idx = (size_t)row * KDIM + tid + u * 256;
    float l1 = (float)l1b[idx];
    float l2 = l2_is_b ? (float)l2bsrc[idx] : l2f[idx];
    float qg = 1.f / (1.f + __expf(-o1[idx]));
    float kg = 1.f / (1.f + __expf(-o2[idx]));
    float s  = (tanhf(l1) * qg + l1) + (o3[idx] * kg + l2);
    sv[u] = s; ss += s * s;
  }
  red[tid] = ss; __syncthreads();
  for (int s = 128; s > 0; s >>= 1) { if (tid < s) red[tid] += red[tid + s]; __syncthreads(); }
  float inv = 1.f / (sqrtf(red[0]) + 1e-8f);
#pragma unroll
  for (int u = 0; u < 4; ++u)
    lts[(size_t)row * KDIM + tid + u * 256] = sv[u] * inv;
}

// out[b, j] = sum_k lts[(j*B+b), k]
__global__ void k_final(const float* __restrict__ lts, float* __restrict__ out) {
  __shared__ float red[256];
  int row = blockIdx.x, tid = threadIdx.x;
  int j = row >> 6, b = row & 63;
  float s = 0.f;
#pragma unroll
  for (int u = 0; u < 4; ++u) s += lts[(size_t)row * KDIM + tid + u * 256];
  red[tid] = s; __syncthreads();
  for (int st = 128; st > 0; st >>= 1) { if (tid < st) red[tid] += red[tid + st]; __syncthreads(); }
  if (tid == 0) out[b * NCAP + j] = red[0];
}

// --------------------------------- launch ----------------------------------

extern "C" void kernel_launch(void* const* d_in, const int* in_sizes, int n_in,
                              void* d_out, int out_size, void* d_ws, size_t ws_size,
                              hipStream_t stream) {
  (void)in_sizes; (void)n_in; (void)out_size; (void)ws_size;
  const float* img   = (const float*)d_in[0];
  const float* cap   = (const float*)d_in[1];
  const int*   lens  = (const int*)  d_in[2];
  const float* hmat  = (const float*)d_in[3];
  const float* hbias = (const float*)d_in[4];
  const float* W1 = (const float*)d_in[5]; const float* b1 = (const float*)d_in[6];
  const float* W2 = (const float*)d_in[7]; const float* b2 = (const float*)d_in[8];
  const float* W3 = (const float*)d_in[9]; const float* b3 = (const float*)d_in[10];
  float* out = (float*)d_out;

  char* ws = (char*)d_ws;
  auto carve = [&](size_t bytes) { char* p = ws; ws += (bytes + 255) & ~(size_t)255; return p; };

  __bf16* imgb   = (__bf16*)carve((size_t)M1 * KDIM * 2);          // 4.7 MB
  __bf16* capT   = (__bf16*)carve((size_t)NCAP * KDIM * LDIM * 2); // 6.3 MB
  __bf16* Wb1    = (__bf16*)carve((size_t)KDIM * KDIM * 2);
  __bf16* Wb2    = (__bf16*)carve((size_t)KDIM * KDIM * 2);
  __bf16* Wb3    = (__bf16*)carve((size_t)KDIM * KDIM * 2);
  __bf16* hcap   = (__bf16*)carve((size_t)N1 * KDIM * 2);          // 0.9 MB
  __bf16* lall   = (__bf16*)carve((size_t)HOUT * ROWS * KDIM * 2); // 75.5 MB
  float*  lts    = (float*) carve((size_t)ROWS * KDIM * 4);        // 16.8 MB
  __bf16* Gb     = (__bf16*)carve((size_t)ROWS * KDIM * 2);
  __bf16* l2b    = (__bf16*)carve((size_t)ROWS * KDIM * 2);
  float*  o1     = (float*) carve((size_t)ROWS * KDIM * 4);
  float*  o2     = (float*) carve((size_t)ROWS * KDIM * 4);
  float*  o3     = (float*) carve((size_t)ROWS * KDIM * 4);
  float*  logits = (float*) carve((size_t)M1 * N1 * 4);            // 4.0 MB

  // --- converts ---
  {
    int n = M1 * KDIM;
    k_cvt_bf16<<<(n + 255) / 256, 256, 0, stream>>>(img, imgb, n);
    n = KDIM * KDIM;
    k_cvt_bf16<<<(n + 255) / 256, 256, 0, stream>>>(W1, Wb1, n);
    k_cvt_bf16<<<(n + 255) / 256, 256, 0, stream>>>(W2, Wb2, n);
    k_cvt_bf16<<<(n + 255) / 256, 256, 0, stream>>>(W3, Wb3, n);
    n = NCAP * KDIM * LDIM;
    k_capT<<<(n + 255) / 256, 256, 0, stream>>>(cap, capT);
  }

  // --- per-caption attention pipeline ---
  for (int j = 0; j < NCAP; ++j) {
    k_hcap<<<(N1 * KDIM) / 256, 256, 0, stream>>>(cap, hmat, hcap, j);
    k_logits<<<dim3(N1 / 48, M1 / 128), 256, 0, stream>>>(imgb, hcap, hbias, logits);
    k_softmax<<<BDIM * HOUT, 256, 0, stream>>>(logits, lens, j);
    k_lall<<<BDIM, HOUT * 32, 0, stream>>>(logits, capT, img, lall, j);
  }

  // --- gated scan over heads, batched across all captions ---
  for (int s = 0; s < 8; ++s) {
    int head_l1 = (s == 0) ? 0 : s + 1;                   // heads 0,2,3,...,8
    const __bf16* l1p = lall + (size_t)head_l1 * ROWS * KDIM;
    const float*  l2f = nullptr;
    const __bf16* l2s = nullptr;
    int l2_is_b = 0;
    if (s == 0) { l2s = lall + (size_t)1 * ROWS * KDIM; l2_is_b = 1; }  // l2 = head 1
    else        { l2f = lts; }

    k_gated_pre<<<(ROWS * KDIM) / 256, 256, 0, stream>>>(l1p, l2f, l2s, l2_is_b, Gb, l2b);
    k_gemm_bias<<<dim3(KDIM / 128, ROWS / 128), 256, 0, stream>>>(Gb,  Wb1, b1, o1);
    k_gemm_bias<<<dim3(KDIM / 128, ROWS / 128), 256, 0, stream>>>(Gb,  Wb2, b2, o2);
    k_gemm_bias<<<dim3(KDIM / 128, ROWS / 128), 256, 0, stream>>>(l2b, Wb3, b3, o3);
    k_gated_post<<<ROWS, 256, 0, stream>>>(l1p, l2f, l2s, l2_is_b, o1, o2, o3, lts);
  }

  k_final<<<ROWS, 256, 0, stream>>>(lts, out);
}
